// MultiHeadAttention_60258391163108
// MI455X (gfx1250) — matrix operand order repaired and checked by reading, not compiled, not run on hardware
//
#include <hip/hip_runtime.h>
#include <hip/hip_bf16.h>

typedef __attribute__((ext_vector_type(16))) __bf16 v16bf;
typedef __attribute__((ext_vector_type(8)))  float  v8f;
typedef __attribute__((ext_vector_type(8)))  __bf16 v8bf;

#define S_LEN  4096
#define DMODEL 512
#define NH     8
#define DK     64

__device__ __forceinline__ v8f wmma_bf16(v16bf a, v16bf b, v8f c) {
  return __builtin_amdgcn_wmma_f32_16x16x32_bf16(false, a, false, b, (short)0, c,
                                                 false, false);
}

// Low 32 bits of a flat pointer into LDS == LDS byte address (ISA 10.2 aperture map).
__device__ __forceinline__ unsigned lds_u32(const void* p) {
  return (unsigned)(unsigned long long)(size_t)p;
}
// CDNA5 async DMA: 16B per lane, LDS destination address taken from a VGPR.
__device__ __forceinline__ void async_b128(unsigned lds_addr, unsigned goff,
                                           const void* base) {
  asm volatile("global_load_async_to_lds_b128 %0, %1, %2"
               :: "v"(lds_addr), "v"(goff), "s"(base) : "memory");
}
__device__ __forceinline__ void wait_async0() {
  asm volatile("s_wait_asynccnt 0x0" ::: "memory");
}

// ============================================================================
// GEMM: Y[m][n] = sum_k X[m][k] * W[n][k] + bias[n]
// Macro tile 128x64, 8 waves, each wave 32x32 (2x2 WMMA frags), K-step 32.
// Double-buffered LDS (single barrier per K-step; next-tile global loads
// overlap current-tile WMMAs).
// OMODE 0: bf16 head-major [B*H][S][DK]      (K path)
// OMODE 1: bf16 head-transposed [B*H][DK][S] (Q,V paths)
// OMODE 2: fp32 row-major [m][n]             (final projection)
// ============================================================================
template <typename TIN, int OMODE>
__global__ void __launch_bounds__(256)
proj_gemm(const TIN* __restrict__ X, const float* __restrict__ W,
          const float* __restrict__ bias, void* __restrict__ OUT) {
  __shared__ __align__(32) char lds[2][12 * 1024];  // per buf: A 8KB, B 4KB

  const int tid   = threadIdx.x;
  const int lane  = tid & 31;
  const int wave  = tid >> 5;
  const int mbase = blockIdx.y * 128;
  const int nbase = blockIdx.x * 64;
  const int wm    = (wave & 3) * 2;
  const int wn    = (wave >> 2) * 2;

  // staging assignments (vectorized: 16B LDS stores)
  const int ma = tid & 127, kh = (tid >> 7) * 16;  // A: row + 16-k half
  const int kB = (tid >> 6) * 8, n8 = tid & 63;    // B: 8-k group + n

  v8f acc[2][2] = {};

  auto stage = [&](int kb, char* buf) {
    // A tile (128x32): 64B contiguous reads, 2x16B LDS stores
    const size_t xrow = (size_t)(mbase + ma) * DMODEL + kb + kh;
    v8bf lo, hi;
#pragma unroll
    for (int j = 0; j < 8; ++j) lo[j] = (__bf16)(float)X[xrow + j];
#pragma unroll
    for (int j = 0; j < 8; ++j) hi[j] = (__bf16)(float)X[xrow + 8 + j];
    char* fr = buf + (ma >> 4) * 1024 + (ma & 15) * 32 + kh;
    *(v8bf*)fr         = lo;   // k-octet -> lanes 0-15 half
    *(v8bf*)(fr + 512) = hi;   // next octet -> lanes 16-31 half
    // B tile: B[k][n]=W[n][k]; reads coalesced along k, one 16B LDS store
    v8bf bv;
#pragma unroll
    for (int j = 0; j < 8; ++j)
      bv[j] = (__bf16)W[(size_t)(nbase + n8) * DMODEL + kb + kB + j];
    *(v8bf*)(buf + 8192 + (n8 >> 4) * 1024 + ((n8 & 15) + ((kB >> 3) & 1) * 16) * 32 + (kB >> 4) * 16) = bv;
  };
  auto compute = [&](const char* buf) {
    v16bf a0 = *(const v16bf*)(buf + (wm + 0) * 1024 + lane * 32);
    v16bf a1 = *(const v16bf*)(buf + (wm + 1) * 1024 + lane * 32);
    v16bf b0 = *(const v16bf*)(buf + 8192 + (wn + 0) * 1024 + lane * 32);
    v16bf b1 = *(const v16bf*)(buf + 8192 + (wn + 1) * 1024 + lane * 32);
    acc[0][0] = wmma_bf16(a0, b0, acc[0][0]);
    acc[0][1] = wmma_bf16(a0, b1, acc[0][1]);
    acc[1][0] = wmma_bf16(a1, b0, acc[1][0]);
    acc[1][1] = wmma_bf16(a1, b1, acc[1][1]);
  };

  stage(0, lds[0]);
  __syncthreads();
  int cur = 0;
  for (int kb = 0; kb < DMODEL; kb += 32) {
    if (kb + 32 < DMODEL) stage(kb + 32, lds[cur ^ 1]);  // overlaps compute
    compute(lds[cur]);
    __syncthreads();
    cur ^= 1;
  }

  // epilogue: C/D layout -> lane holds column n = lane&15, rows r + 8*(lane>>4)
  const int half = lane >> 4;
  const int nl   = lane & 15;
#pragma unroll
  for (int i = 0; i < 2; ++i)
#pragma unroll
    for (int j = 0; j < 2; ++j) {
      int   n  = nbase + (wn + j) * 16 + nl;
      float bv = bias[n];
      int   m0 = mbase + (wm + i) * 16 + 8 * half;  // 8 consecutive rows
      if (OMODE == 1) {  // transposed: contiguous along t -> one 16B store
        int  bb = m0 >> 12, t0 = m0 & 4095;
        int  h = n >> 6, d = n & 63;
        v8bf o;
#pragma unroll
        for (int r = 0; r < 8; ++r) o[r] = (__bf16)(acc[i][j][r] + bv);
        *(v8bf*)((__bf16*)OUT + ((size_t)(bb * NH + h) * DK + d) * S_LEN + t0) = o;
      } else {
#pragma unroll
        for (int r = 0; r < 8; ++r) {
          int   m = m0 + r;
          float v = acc[i][j][r] + bv;
          if (OMODE == 0) {
            int bb = m >> 12, t = m & 4095;
            int h = n >> 6, d = n & 63;
            ((__bf16*)OUT)[((size_t)(bb * NH + h) * S_LEN + t) * DK + d] = (__bf16)v;
          } else {
            ((float*)OUT)[(size_t)m * DMODEL + n] = v;
          }
        }
      }
    }
}

// ============================================================================
// Flash attention over one (b,h): 128 q rows per workgroup (8 waves x 16 q).
// 64-key tiles, double-buffered via async global->LDS b128 DMA (ASYNCcnt),
// key loop manually unrolled x2 over the two buffers.
// S^T = K_tile x Q^T (key reduction lane-local + shfl_xor 16), online softmax,
// O^T += V^T x P^T.  K is [bh][t][64]; Q^T,V^T are [bh][64][4096].
// ============================================================================
__global__ void __launch_bounds__(256)
flash_attn(const __bf16* __restrict__ Qt, const __bf16* __restrict__ Kh,
           const __bf16* __restrict__ Vt, __bf16* __restrict__ Out) {
  __shared__ __align__(32) char lds[48 * 1024];
  // buf b at lds + b*16384: K frags [0,8KB), V^T frags [8KB,16KB).
  // P at 32KB + wave*2KB (64 keys x 16 q bf16, B-frag order).
  char* ldsP = lds + 32 * 1024;

  const int tid   = threadIdx.x;
  const int lane  = tid & 31;
  const int wave  = tid >> 5;
  const int bh    = blockIdx.y;
  const int qbase = blockIdx.x * 128 + wave * 16;
  const int half  = lane >> 4;

  const __bf16* qt = Qt + (size_t)bh * DK * S_LEN;
  const __bf16* kg = Kh + (size_t)bh * S_LEN * DK;
  const __bf16* vt = Vt + (size_t)bh * DK * S_LEN;

  // per-thread async-DMA slots: 2x16B for K, 2x16B for V per 64-key tile.
  // Fragment permutation is encoded in the per-lane LDS destination address.
  const int keyt = tid >> 3, d8 = (tid & 7) * 8;  // K tile: 64 keys x 64 d
  const int dv   = tid >> 2, k8 = (tid & 3) * 8;  // V^T tile: 64 d x 64 keys
  unsigned kofK[2], kofV[2];
#pragma unroll
  for (int c = 0; c < 2; ++c) {
    int kc = keyt + 32 * c;  // K frag = (key>>4)*2 + (d>>5)
    kofK[c] = (unsigned)(((kc >> 4) * 2 + (d8 >> 5)) * 1024 +
                         ((kc & 15) + (((d8 >> 3) & 1) << 4)) * 32 +
                         (((d8 >> 4) & 1) << 3) * 2);
    // V frag = (d>>4)*2 + (key>>5); key-local = k8
    kofV[c] = (unsigned)(8192 + ((dv >> 4) * 2 + c) * 1024 +
                         ((dv & 15) + (((k8 >> 3) & 1) << 4)) * 32 +
                         ((k8 >> 4) << 3) * 2);
  }
  const unsigned ldsbase = lds_u32(lds);

  auto prefetch = [&](int kb, unsigned nb) {
#pragma unroll
    for (int c = 0; c < 2; ++c) {
      async_b128(nb + kofK[c], (unsigned)(((kb + keyt + 32 * c) * DK + d8) * 2), kg);
      async_b128(nb + kofV[c], (unsigned)((dv * S_LEN + kb + k8 + 32 * c) * 2), vt);
    }
  };

  // Q^T B-fragments (d x q) straight from transposed global, fold 1/sqrt(dk)
  v16bf qb[2];
#pragma unroll
  for (int f = 0; f < 2; ++f) {
#pragma unroll
    for (int e = 0; e < 16; ++e)
      qb[f][e] = (__bf16)((float)qt[(size_t)(f * 32 + 8 * half + (e & 7) + ((e >> 3) << 4)) * S_LEN + qbase + (lane & 15)] * 0.125f);
  }

  v8f   acc[4] = {};
  float m_i = -__builtin_inff();
  float l_i = 0.0f;

  auto compute = [&](const char* bK) {
    const char* bV = bK + 8192;
    v8f s[4];
#pragma unroll
    for (int t = 0; t < 4; ++t) {  // 16-key blocks
      v16bf k0 = *(const v16bf*)(bK + (t * 2 + 0) * 1024 + lane * 32);
      v16bf k1 = *(const v16bf*)(bK + (t * 2 + 1) * 1024 + lane * 32);
      v8f z = {};
      s[t] = wmma_bf16(k0, qb[0], z);
      s[t] = wmma_bf16(k1, qb[1], s[t]);
    }
    // online softmax; lane holds q = lane&15, keys {t*16 + r + 8*half}
    float tm = -__builtin_inff();
#pragma unroll
    for (int t = 0; t < 4; ++t)
#pragma unroll
      for (int r = 0; r < 8; ++r) tm = fmaxf(tm, s[t][r]);
    tm = fmaxf(tm, __shfl_xor(tm, 16, 32));
    float m_new = fmaxf(m_i, tm);
    float corr  = __expf(m_i - m_new);
    float rs    = 0.0f;
    char* pw = ldsP + wave * 2048;
    int   q0 = lane & 15;
#pragma unroll
    for (int t = 0; t < 4; ++t)
#pragma unroll
      for (int r = 0; r < 8; ++r) {
        float p = __expf(s[t][r] - m_new);
        rs += p;
        int key = t * 16 + r + 8 * half;
        *(__bf16*)(pw + (key >> 5) * 1024 + (q0 + ((key >> 3) & 1) * 16) * 32 + ((key >> 4) & 1) * 16 + (key & 7) * 2) = (__bf16)p;
      }
    rs += __shfl_xor(rs, 16, 32);
    l_i = l_i * corr + rs;
    m_i = m_new;
#pragma unroll
    for (int j = 0; j < 4; ++j)
#pragma unroll
      for (int r = 0; r < 8; ++r) acc[j][r] *= corr;

    asm volatile("s_wait_dscnt 0x0" ::: "memory");  // P^T write->read, same wave
    v16bf pb0 = *(const v16bf*)(pw + lane * 32);           // keys 0..31
    v16bf pb1 = *(const v16bf*)(pw + 1024 + lane * 32);    // keys 32..63
#pragma unroll
    for (int j = 0; j < 4; ++j) {  // d-blocks
      v16bf v0 = *(const v16bf*)(bV + (j * 2 + 0) * 1024 + lane * 32);
      v16bf v1 = *(const v16bf*)(bV + (j * 2 + 1) * 1024 + lane * 32);
      acc[j] = wmma_bf16(v0, pb0, acc[j]);
      acc[j] = wmma_bf16(v1, pb1, acc[j]);
    }
  };

  prefetch(0, ldsbase);
  for (int kb = 0; kb < S_LEN; kb += 128) {
    wait_async0();      // buffer0 tile landed (my wave)
    __syncthreads();    // ... for all waves; buffer1 readers done
    prefetch(kb + 64, ldsbase + 16384);   // always valid: kb+64 <= 4032
    compute(lds);
    wait_async0();      // buffer1 tile landed
    __syncthreads();
    if (kb + 128 < S_LEN) prefetch(kb + 128, ldsbase);
    compute(lds + 16384);
  }

  // writeback: O[q][d] = acc[j][r]/l ; per lane q fixed, d contiguous -> 16B stores
  float inv  = 1.0f / l_i;
  int   bb   = bh >> 3, h = bh & 7;
  int   qrow = qbase + (lane & 15);
#pragma unroll
  for (int j = 0; j < 4; ++j) {
    v8bf o;
#pragma unroll
    for (int r = 0; r < 8; ++r) o[r] = (__bf16)(acc[j][r] * inv);
    int d = j * 16 + 8 * half;
    *(v8bf*)(Out + ((size_t)(bb * S_LEN + qrow)) * DMODEL + h * DK + d) = o;
  }
}

// ============================================================================
extern "C" void kernel_launch(void* const* d_in, const int* in_sizes, int n_in,
                              void* d_out, int out_size, void* d_ws, size_t ws_size,
                              hipStream_t stream) {
  const float* q   = (const float*)d_in[0];
  const float* kk  = (const float*)d_in[1];
  const float* vv  = (const float*)d_in[2];
  const float* w_q = (const float*)d_in[3];
  const float* b_q = (const float*)d_in[4];
  const float* w_k = (const float*)d_in[5];
  const float* b_k = (const float*)d_in[6];
  const float* w_v = (const float*)d_in[7];
  const float* b_v = (const float*)d_in[8];
  const float* w_o = (const float*)d_in[9];
  const float* b_o = (const float*)d_in[10];

  const size_t plane = (size_t)2 * S_LEN * DMODEL;  // elements per bf16 plane
  __bf16* qt = (__bf16*)d_ws;      // [bh][64][4096]
  __bf16* kh = qt + plane;         // [bh][4096][64]
  __bf16* vt = kh + plane;         // [bh][64][4096]
  __bf16* ao = vt + plane;         // [b][t][512]

  dim3 blk(256);
  dim3 pgrid(DMODEL / 64, (2 * S_LEN) / 128);  // (8, 64)
  proj_gemm<float, 1><<<pgrid, blk, 0, stream>>>(q,  w_q, b_q, qt);
  proj_gemm<float, 0><<<pgrid, blk, 0, stream>>>(kk, w_k, b_k, kh);
  proj_gemm<float, 1><<<pgrid, blk, 0, stream>>>(vv, w_v, b_v, vt);

  flash_attn<<<dim3(S_LEN / 128, 2 * NH), blk, 0, stream>>>(qt, kh, vt, ao);

  proj_gemm<__bf16, 2><<<pgrid, blk, 0, stream>>>(ao, w_o, b_o, d_out);
}
